// PointNet2Encoder_71700184040087
// MI455X (gfx1250) — matrix operand order repaired and checked by reading, not compiled, not run on hardware
//
#include <hip/hip_runtime.h>
#include <hip/hip_bf16.h>

typedef __attribute__((ext_vector_type(16))) _Float16 v16h;
typedef __attribute__((ext_vector_type(8)))  float    v8f;
typedef unsigned int __attribute__((ext_vector_type(4))) u32x4;
typedef int          __attribute__((ext_vector_type(8))) i32x8;
typedef int          __attribute__((ext_vector_type(4))) i32x4;

#define DEVINL __device__ __forceinline__

// ---------------------------------------------------------------------------
// WMMA fragment loaders (contiguous k = 16*laneHi + s for A and B, not the ISA table order)
// ---------------------------------------------------------------------------

// A: 16x32 f16 tile, row-major in LDS with leading dim lda (halves).
// lanes 0-15 : row = lane,    K = k0+0..15
// lanes 16-31: row = lane-16, K = k0+16..31
DEVINL v16h load_A_frag(const _Float16* As, int lda, int m0, int k0, int lane) {
  const int row = m0 + (lane & 15);
  const int hi  = (lane >> 4) & 1;
  const _Float16* p = As + (size_t)row * lda + k0 + hi * 16;
  v16h a;
#pragma unroll
  for (int i = 0; i < 8; ++i) { a[i] = p[i]; a[i + 8] = p[i + 8]; }
  return a;
}

// B: 32x16 f16, weights stored transposed/N-major: Wt[n][kpad], kpad % 32 == 0.
// lanes 0-15 : col = lane,    K = k0+0..15   (contiguous, 32B aligned)
// lanes 16-31: col = lane-16, K = k0+16..31
DEVINL v16h load_B_frag(const _Float16* Wt, int kpad, int n0, int k0, int lane) {
  const _Float16* p = Wt + (size_t)(n0 + (lane & 15)) * kpad + k0 + ((lane >> 4) & 1) * 16;
  return *(const v16h*)p;  // 32-byte aligned by construction
}

// ---------------------------------------------------------------------------
// Fused tile GEMM: C = A @ W^T + bias, with epilogues:
// MODE 0: relu -> f16 LDS tile Bs (ldb)
// MODE 1: relu -> column max into LDS maxbuf (uint bits; values >= 0)
// MODE 2: no relu -> f32 global out (ldo)
// blockDim.x == 256 (8 waves). D layout: VGPR i -> row i + 8*laneHi, col lane&15.
// ---------------------------------------------------------------------------
template <int MODE>
DEVINL void wmma_gemm(const _Float16* As, int lda, int MT,
                      const _Float16* Wt, int kpad, int NT,
                      const float* __restrict__ bias,
                      _Float16* Bs, int ldb,
                      unsigned* maxbuf,
                      float* gout, int ldo,
                      int tid) {
  const int wave = tid >> 5, lane = tid & 31;
  const int nk   = kpad >> 5;
  const int row0 = ((lane >> 4) & 1) * 8;
  const int col  = lane & 15;
  for (int t = wave; t < MT * NT; t += 8) {
    const int mt = t / NT, nt = t - mt * NT;
    v8f acc = {};
    for (int kk = 0; kk < nk; ++kk) {
      v16h a = load_A_frag(As, lda, mt * 16, kk * 32, lane);
      v16h b = load_B_frag(Wt, kpad, nt * 16, kk * 32, lane);
      acc = __builtin_amdgcn_wmma_f32_16x16x32_f16(false, a, false, b,
                                                   (short)0, acc, false, false);
    }
    const int n  = nt * 16 + col;
    const float bn = bias[n];
#pragma unroll
    for (int i = 0; i < 8; ++i) {
      const int m = mt * 16 + row0 + i;
      float v = acc[i] + bn;
      if (MODE == 0) {
        v = v > 0.0f ? v : 0.0f;
        Bs[(size_t)m * ldb + n] = (_Float16)v;
      } else if (MODE == 1) {
        v = v > 0.0f ? v : 0.0f;
        atomicMax(&maxbuf[n], __float_as_uint(v));
      } else {
        gout[(size_t)m * ldo + n] = v;
      }
    }
  }
}

// ---------------------------------------------------------------------------
// TDM: async 2D f16 tile copy global -> LDS (CDNA5 ISA).
// D# group0/group1 packed per the ISA bitfield tables; groups 2/3 unused (2D).
// This toolchain exposes the 6-arg builtin (extra i32x8, zero-filled).
// Issue from one wave; follow with s_wait_tensorcnt + workgroup barrier.
// ---------------------------------------------------------------------------
DEVINL void tdm_load_2d_f16(const _Float16* gsrc, unsigned lds_byte_off,
                            unsigned width_elems, unsigned rows,
                            unsigned gstride_elems,
                            unsigned pad_enable, unsigned pad_interval,
                            unsigned pad_amount) {
  const unsigned long long ga = (unsigned long long)(size_t)gsrc;
  u32x4 g0;
  g0[0] = 1u;                                             // count=1 (valid user D#)
  g0[1] = lds_byte_off;                                   // lds_addr [63:32]
  g0[2] = (unsigned)(ga & 0xFFFFFFFFu);                   // global_addr lo
  g0[3] = (unsigned)((ga >> 32) & 0x01FFFFFFu) | (2u << 30);  // addr hi | type=2
  i32x8 g1;
  g1[0] = (int)((1u << 16) |                              // data_size = 2B
                (pad_enable << 20) |
                (pad_interval << 22) |
                (pad_amount << 25));
  g1[1] = (int)((width_elems & 0xFFFFu) << 16);           // tensor_dim0[15:0]
  g1[2] = (int)(((width_elems >> 16) & 0xFFFFu) |         // tensor_dim0[31:16]
                ((rows & 0xFFFFu) << 16));                // tensor_dim1[15:0]
  g1[3] = (int)(((rows >> 16) & 0xFFFFu) |                // tensor_dim1[31:16]
                ((width_elems & 0xFFFFu) << 16));         // tile_dim0
  g1[4] = (int)(rows & 0xFFFFu);                          // tile_dim1 ; tile_dim2=0
  g1[5] = (int)gstride_elems;                             // tensor_dim0_stride lo
  g1[6] = 0;                                              // stride hi ; dim1_stride lo
  g1[7] = 0;
  i32x4 z4 = {};
  i32x8 z8 = {};
  __builtin_amdgcn_tensor_load_to_lds(g0, g1, z4, z4, z8, 0);
}

DEVINL unsigned lds_off_of(const void* p) {
  // addrspace(3) offsets live in the low 32 bits of the flat LDS aperture addr
  return (unsigned)(size_t)p;
}

// ---------------------------------------------------------------------------
// Ball query: first `nsample` indices (in index order) with dist^2 <= r2.
// Wave 0 performs a ballot-compaction scan; result in LDS nbr[nsample].
// ---------------------------------------------------------------------------
DEVINL void ball_query(const float* pts, int n, float cx, float cy, float cz,
                       float r2, int nsample, int* nbr, int tid) {
  if (tid < 32) {
    const int lane = tid;
    int count = 0, first = 0;
    bool haveFirst = false;
    for (int base = 0; base < n && count < nsample; base += 32) {
      const int i = base + lane;
      bool in = false;
      if (i < n) {
        float dx = pts[i * 3 + 0] - cx;
        float dy = pts[i * 3 + 1] - cy;
        float dz = pts[i * 3 + 2] - cz;
        in = (dx * dx + dy * dy + dz * dz) <= r2;
      }
      unsigned mask = (unsigned)__builtin_amdgcn_ballot_w32(in);
      if (!haveFirst && mask) { first = base + __builtin_ctz(mask); haveFirst = true; }
      if (in) {
        int pos = count + __builtin_popcount(mask & ((1u << lane) - 1u));
        if (pos < nsample) nbr[pos] = i;
      }
      count += __builtin_popcount(mask);
    }
    if (count > nsample) count = nsample;
    for (int j = count + lane; j < nsample; j += 32) nbr[j] = first;
  }
  __syncthreads();
}

// ---------------------------------------------------------------------------
// Weight prep: fold BN (scale = gamma/sqrt(1+eps)), transpose to N-major,
// cast f16, zero-pad K to kpad.  Wt[n*kpad + k], bOut[n].
// feats_first: permute K so source rows [3..cin) come first, xyz rows last
// (matches TDM-friendly features-first A-tile layout).
// ---------------------------------------------------------------------------
__global__ void prep_weights(const float* __restrict__ W, const float* __restrict__ b,
                             const float* __restrict__ gamma, const float* __restrict__ beta,
                             int cin, int cout, int kpad, int feats_first,
                             _Float16* __restrict__ Wt, float* __restrict__ bOut) {
  const float inv_std = 0.9999950000374997f;  // 1/sqrt(1+1e-5)
  const int idx = blockIdx.x * blockDim.x + threadIdx.x;
  if (idx < cout * kpad) {
    const int n = idx / kpad, k = idx - n * kpad;
    int src;
    if (feats_first) src = (k < cin - 3) ? (k + 3) : ((k < cin) ? (k - (cin - 3)) : -1);
    else             src = (k < cin) ? k : -1;
    const float s = gamma ? gamma[n] * inv_std : 1.0f;
    const float v = (src >= 0) ? W[(size_t)src * cout + n] * s : 0.0f;
    Wt[idx] = (_Float16)v;
  }
  if (idx < cout) {
    const float s = gamma ? gamma[idx] * inv_std : 1.0f;
    bOut[idx] = b[idx] * s + (beta ? beta[idx] : 0.0f);
  }
}

// ---------------------------------------------------------------------------
// Farthest point sampling: one block per batch; n == PER*256.
// ---------------------------------------------------------------------------
template <int PER>
__global__ void fps_kernel(const float* __restrict__ xyz, int n, int m,
                           float* __restrict__ outXyz) {
  __shared__ float sval[256];
  __shared__ int   sidx[256];
  const int b = blockIdx.x, tid = threadIdx.x;
  const float* p = xyz + (size_t)b * n * 3;
  float dist[PER];
#pragma unroll
  for (int j = 0; j < PER; ++j) dist[j] = 1e10f;
  int far = 0;
  for (int it = 0; it < m; ++it) {
    const float fx = p[far * 3 + 0], fy = p[far * 3 + 1], fz = p[far * 3 + 2];
    if (tid == 0) {
      float* o = outXyz + ((size_t)b * m + it) * 3;
      o[0] = fx; o[1] = fy; o[2] = fz;
    }
    float best = -1.0f; int bi = 0;
#pragma unroll
    for (int j = 0; j < PER; ++j) {
      const int i = j * 256 + tid;
      const float dx = p[i * 3 + 0] - fx, dy = p[i * 3 + 1] - fy, dz = p[i * 3 + 2] - fz;
      const float d = dx * dx + dy * dy + dz * dz;
      if (d < dist[j]) dist[j] = d;
      if (dist[j] > best) { best = dist[j]; bi = i; }
    }
    sval[tid] = best; sidx[tid] = bi;
    __syncthreads();
    for (int s = 128; s > 0; s >>= 1) {
      if (tid < s && sval[tid + s] > sval[tid]) { sval[tid] = sval[tid + s]; sidx[tid] = sidx[tid + s]; }
      __syncthreads();
    }
    far = sidx[0];
    __syncthreads();
  }
}

// ---------------------------------------------------------------------------
// SA1: one block per (b, center) of 16x512. grouped 32x3 -> 128 -> 256 -> max.
// l1p output stored f16 (consumed only as f16 WMMA A-fragments).
// ---------------------------------------------------------------------------
__global__ void sa1_kernel(const float* __restrict__ xyz, const float* __restrict__ centers,
                           const _Float16* __restrict__ w1, const float* __restrict__ b1,
                           const _Float16* __restrict__ w2, const float* __restrict__ b2,
                           _Float16* __restrict__ l1p) {
  extern __shared__ char smem[];
  _Float16* A0 = (_Float16*)smem;                                 // 32 x 32
  _Float16* A1 = (_Float16*)(smem + 32 * 32 * 2);                 // 32 x 128
  unsigned* maxbuf = (unsigned*)(smem + 32 * 32 * 2 + 32 * 128 * 2);  // 256
  int* nbr = (int*)(maxbuf + 256);                                // 32
  const int tid = threadIdx.x;
  const int b = blockIdx.x >> 9, c = blockIdx.x & 511;
  const float* pts = xyz + (size_t)b * 8192 * 3;
  const float* ctr = centers + ((size_t)b * 512 + c) * 3;
  const float cx = ctr[0], cy = ctr[1], cz = ctr[2];
  ball_query(pts, 8192, cx, cy, cz, 0.01f, 32, nbr, tid);
  for (int e = tid; e < 32 * 32; e += 256) {
    const int j = e >> 5, k = e & 31;
    float v = 0.0f;
    if (k < 3) v = pts[nbr[j] * 3 + k] - (k == 0 ? cx : (k == 1 ? cy : cz));
    A0[e] = (_Float16)v;
  }
  for (int e = tid; e < 256; e += 256) maxbuf[e] = 0u;
  __syncthreads();
  wmma_gemm<0>(A0, 32, 2, w1, 32, 8, b1, A1, 128, nullptr, nullptr, 0, tid);
  __syncthreads();
  wmma_gemm<1>(A1, 128, 2, w2, 128, 16, b2, nullptr, 0, maxbuf, nullptr, 0, tid);
  __syncthreads();
  _Float16* out = l1p + ((size_t)b * 512 + c) * 256;
  for (int n = tid; n < 256; n += 256) out[n] = (_Float16)__uint_as_float(maxbuf[n]);
}

// ---------------------------------------------------------------------------
// SA2: one block per (b, center) of 16x128. grouped 64x259 -> 256 -> 512 -> max.
// Neighbor features gathered f16 -> f16 (pure copy); l2p output f16.
// ---------------------------------------------------------------------------
__global__ void sa2_kernel(const float* __restrict__ l1x, const _Float16* __restrict__ l1p,
                           const float* __restrict__ centers2,
                           const _Float16* __restrict__ w1, const float* __restrict__ b1,
                           const _Float16* __restrict__ w2, const float* __restrict__ b2,
                           _Float16* __restrict__ l2p, float* __restrict__ l2x_out) {
  extern __shared__ char smem[];
  _Float16* A0 = (_Float16*)smem;                                   // 64 x 288
  _Float16* A1 = (_Float16*)(smem + 64 * 288 * 2);                  // 64 x 256
  unsigned* maxbuf = (unsigned*)(smem + 64 * 288 * 2 + 64 * 256 * 2);  // 512
  int* nbr = (int*)(maxbuf + 512);                                  // 64
  const int tid = threadIdx.x;
  const int b = blockIdx.x >> 7, c = blockIdx.x & 127;
  const float* pts = l1x + (size_t)b * 512 * 3;
  const float* ctr = centers2 + ((size_t)b * 128 + c) * 3;
  const float cx = ctr[0], cy = ctr[1], cz = ctr[2];
  if (tid < 3) l2x_out[((size_t)b * 128 + c) * 3 + tid] = ctr[tid];
  ball_query(pts, 512, cx, cy, cz, 0.04f, 64, nbr, tid);
  const _Float16* feats = l1p + (size_t)b * 512 * 256;
  for (int e = tid; e < 64 * 288; e += 256) {
    const int j = e / 288, k = e - j * 288;
    const int i = nbr[j];
    _Float16 v = (_Float16)0.0f;
    if (k < 3)        v = (_Float16)(pts[i * 3 + k] - (k == 0 ? cx : (k == 1 ? cy : cz)));
    else if (k < 259) v = feats[(size_t)i * 256 + (k - 3)];
    A0[e] = v;
  }
  for (int e = tid; e < 512; e += 256) maxbuf[e] = 0u;
  __syncthreads();
  wmma_gemm<0>(A0, 288, 4, w1, 288, 16, b1, A1, 256, nullptr, nullptr, 0, tid);
  __syncthreads();
  wmma_gemm<1>(A1, 256, 4, w2, 256, 32, b2, nullptr, 0, maxbuf, nullptr, 0, tid);
  __syncthreads();
  _Float16* out = l2p + ((size_t)b * 128 + c) * 512;
  for (int n = tid; n < 512; n += 256) out[n] = (_Float16)__uint_as_float(maxbuf[n]);
}

// ---------------------------------------------------------------------------
// SA3 (global grouping) + global head: one block per batch.
// A0 layout is features-first: [l2p feats(512) | xyz(3) | pad(29)], lda=544.
// Feature block staged by TDM with LDS row padding (1024B rows + 64B pad).
// 128x515 -> 512 -> 1024 -> max over 128 -> gh1(relu) -> gh2.
// ---------------------------------------------------------------------------
__global__ void sa3_kernel(const float* __restrict__ l2x, const _Float16* __restrict__ l2p,
                           const _Float16* __restrict__ w1, const float* __restrict__ b1,
                           const _Float16* __restrict__ w2, const float* __restrict__ b2,
                           const float* __restrict__ gh1W, const float* __restrict__ gh1b,
                           const float* __restrict__ gh2W, const float* __restrict__ gh2b,
                           float* __restrict__ gout) {
  extern __shared__ char smem[];
  _Float16* A0 = (_Float16*)smem;                                    // 128 x 544
  _Float16* A1 = (_Float16*)(smem + 128 * 544 * 2);                  // 128 x 512
  unsigned* maxbuf = (unsigned*)(smem + 128 * 544 * 2 + 128 * 512 * 2);  // 1024
  float* gbuf = (float*)(maxbuf + 1024);                             // 512
  const int tid = threadIdx.x, b = blockIdx.x;
  const float* px = l2x + (size_t)b * 128 * 3;
  const _Float16* pf = l2p + (size_t)b * 128 * 512;
  // Async tensor DMA: 128 rows x 512 f16, global stride 512, LDS stride 544
  // (pad_interval 7 => pad after every 256 DWORDs; pad_amount 15 => 16 DWORDs).
  if (tid < 32) {
    tdm_load_2d_f16(pf, lds_off_of(A0), 512u, 128u, 512u, 1u, 7u, 15u);
    __builtin_amdgcn_s_wait_tensorcnt(0);
  }
  // Tail columns 512..543: xyz then zero pad (disjoint from TDM target bytes).
  for (int e = tid; e < 128 * 32; e += 256) {
    const int j = e >> 5, k = 512 + (e & 31);
    _Float16 v = (_Float16)0.0f;
    if (k < 515) v = (_Float16)px[j * 3 + (k - 512)];
    A0[(size_t)j * 544 + k] = v;
  }
  for (int e = tid; e < 1024; e += 256) maxbuf[e] = 0u;
  __syncthreads();
  wmma_gemm<0>(A0, 544, 8, w1, 544, 32, b1, A1, 512, nullptr, nullptr, 0, tid);
  __syncthreads();
  wmma_gemm<1>(A1, 512, 8, w2, 512, 64, b2, nullptr, 0, maxbuf, nullptr, 0, tid);
  __syncthreads();
  // g = relu(l3p @ gh1W + b)  (M=1: plain VALU dot products)
  for (int n = tid; n < 512; n += 256) {
    float acc = gh1b[n];
    for (int k = 0; k < 1024; ++k) acc += __uint_as_float(maxbuf[k]) * gh1W[(size_t)k * 512 + n];
    gbuf[n] = acc > 0.0f ? acc : 0.0f;
  }
  __syncthreads();
  for (int n = tid; n < 256; n += 256) {
    float acc = gh2b[n];
    for (int k = 0; k < 512; ++k) acc += gbuf[k] * gh2W[(size_t)k * 256 + n];
    gout[(size_t)b * 256 + n] = acc;
  }
}

// ---------------------------------------------------------------------------
// Point head: (B*128) x 512 -> relu 256 -> 256. 32 rows per block.
// A0 is a contiguous 32x512 f16 tile: single TDM copy, no padding.
// ---------------------------------------------------------------------------
__global__ void ph_kernel(const _Float16* __restrict__ l2p,
                          const _Float16* __restrict__ w1, const float* __restrict__ b1,
                          const _Float16* __restrict__ w2, const float* __restrict__ b2,
                          float* __restrict__ pout) {
  extern __shared__ char smem[];
  _Float16* A0 = (_Float16*)smem;                   // 32 x 512
  _Float16* A1 = (_Float16*)(smem + 32 * 512 * 2);  // 32 x 256
  const int tid = threadIdx.x;
  const int r0 = blockIdx.x * 32;
  if (tid < 32) {
    tdm_load_2d_f16(l2p + (size_t)r0 * 512, lds_off_of(A0), 512u, 32u, 512u, 0u, 0u, 0u);
    __builtin_amdgcn_s_wait_tensorcnt(0);
  }
  __syncthreads();
  wmma_gemm<0>(A0, 512, 2, w1, 512, 16, b1, A1, 256, nullptr, nullptr, 0, tid);
  __syncthreads();
  wmma_gemm<2>(A1, 256, 2, w2, 256, 16, b2, nullptr, 0, nullptr,
               pout + (size_t)r0 * 256, 256, tid);
}

// ---------------------------------------------------------------------------
// Host launch
// ---------------------------------------------------------------------------
static inline size_t align256(size_t x) { return (x + 255) & ~(size_t)255; }

extern "C" void kernel_launch(void* const* d_in, const int* in_sizes, int n_in,
                              void* d_out, int out_size, void* d_ws, size_t ws_size,
                              hipStream_t stream) {
  (void)in_sizes; (void)n_in; (void)out_size; (void)ws_size;
  // jax tree-flatten order: point_cloud, then params dict sorted by key;
  // linear {'W','b'} -> W,b ; convbn {'W','b','beta','gamma'} -> W,b,beta,gamma.
  const float* pc    = (const float*)d_in[0];
  const float* gh1W  = (const float*)d_in[1];
  const float* gh1b  = (const float*)d_in[2];
  const float* gh2W  = (const float*)d_in[3];
  const float* gh2b  = (const float*)d_in[4];
  const float* ph1W  = (const float*)d_in[5];
  const float* ph1b  = (const float*)d_in[6];
  const float* ph2W  = (const float*)d_in[7];
  const float* ph2b  = (const float*)d_in[8];
  const float* s1aW  = (const float*)d_in[9];
  const float* s1ab  = (const float*)d_in[10];
  const float* s1aBe = (const float*)d_in[11];
  const float* s1aGa = (const float*)d_in[12];
  const float* s1bW  = (const float*)d_in[13];
  const float* s1bb  = (const float*)d_in[14];
  const float* s1bBe = (const float*)d_in[15];
  const float* s1bGa = (const float*)d_in[16];
  const float* s2aW  = (const float*)d_in[17];
  const float* s2ab  = (const float*)d_in[18];
  const float* s2aBe = (const float*)d_in[19];
  const float* s2aGa = (const float*)d_in[20];
  const float* s2bW  = (const float*)d_in[21];
  const float* s2bb  = (const float*)d_in[22];
  const float* s2bBe = (const float*)d_in[23];
  const float* s2bGa = (const float*)d_in[24];
  const float* s3aW  = (const float*)d_in[25];
  const float* s3ab  = (const float*)d_in[26];
  const float* s3aBe = (const float*)d_in[27];
  const float* s3aGa = (const float*)d_in[28];
  const float* s3bW  = (const float*)d_in[29];
  const float* s3bb  = (const float*)d_in[30];
  const float* s3bBe = (const float*)d_in[31];
  const float* s3bGa = (const float*)d_in[32];

  char* ws = (char*)d_ws;
  size_t off = 0;
  auto alloc = [&](size_t bytes) -> char* {
    char* p = ws + off;
    off = align256(off + bytes);
    return p;
  };
  float*    centers1 = (float*)alloc((size_t)16 * 512 * 3 * 4);
  _Float16* l1p      = (_Float16*)alloc((size_t)16 * 512 * 256 * 2);
  float*    centers2 = (float*)alloc((size_t)16 * 128 * 3 * 4);
  _Float16* l2p      = (_Float16*)alloc((size_t)16 * 128 * 512 * 2);
  _Float16* wt1a = (_Float16*)alloc(128 * 32 * 2);    float* bf1a = (float*)alloc(128 * 4);
  _Float16* wt1b = (_Float16*)alloc(256 * 128 * 2);   float* bf1b = (float*)alloc(256 * 4);
  _Float16* wt2a = (_Float16*)alloc(256 * 288 * 2);   float* bf2a = (float*)alloc(256 * 4);
  _Float16* wt2b = (_Float16*)alloc(512 * 256 * 2);   float* bf2b = (float*)alloc(512 * 4);
  _Float16* wt3a = (_Float16*)alloc(512 * 544 * 2);   float* bf3a = (float*)alloc(512 * 4);
  _Float16* wt3b = (_Float16*)alloc(1024 * 512 * 2);  float* bf3b = (float*)alloc(1024 * 4);
  _Float16* wtp1 = (_Float16*)alloc(256 * 512 * 2);   float* bfp1 = (float*)alloc(256 * 4);
  _Float16* wtp2 = (_Float16*)alloc(256 * 256 * 2);   float* bfp2 = (float*)alloc(256 * 4);

  auto prep = [&](const float* W, const float* b, const float* g, const float* be,
                  int cin, int cout, int kpad, int ff, _Float16* Wt, float* bo) {
    const int total = cout * kpad;
    prep_weights<<<(total + 255) / 256, 256, 0, stream>>>(W, b, g, be, cin, cout, kpad, ff, Wt, bo);
  };
  prep(s1aW, s1ab, s1aGa, s1aBe,   3,  128,  32, 0, wt1a, bf1a);
  prep(s1bW, s1bb, s1bGa, s1bBe, 128,  256, 128, 0, wt1b, bf1b);
  prep(s2aW, s2ab, s2aGa, s2aBe, 259,  256, 288, 0, wt2a, bf2a);
  prep(s2bW, s2bb, s2bGa, s2bBe, 256,  512, 256, 0, wt2b, bf2b);
  prep(s3aW, s3ab, s3aGa, s3aBe, 515,  512, 544, 1, wt3a, bf3a);  // feats-first
  prep(s3bW, s3bb, s3bGa, s3bBe, 512, 1024, 512, 0, wt3b, bf3b);
  prep(ph1W, ph1b, nullptr, nullptr, 512, 256, 512, 0, wtp1, bfp1);
  prep(ph2W, ph2b, nullptr, nullptr, 256, 256, 256, 0, wtp2, bfp2);

  float* out_gf  = (float*)d_out;                       // (16,256)
  float* out_pf  = out_gf + 16 * 256;                   // (16,128,256)
  float* out_l2x = out_pf + (size_t)16 * 128 * 256;     // (16,128,3)

  fps_kernel<32><<<16, 256, 0, stream>>>(pc, 8192, 512, centers1);

  const size_t smem1 = 32 * 32 * 2 + 32 * 128 * 2 + 256 * 4 + 32 * 4;
  sa1_kernel<<<16 * 512, 256, smem1, stream>>>(pc, centers1, wt1a, bf1a, wt1b, bf1b, l1p);

  fps_kernel<2><<<16, 256, 0, stream>>>(centers1, 512, 128, centers2);

  const size_t smem2 = 64 * 288 * 2 + 64 * 256 * 2 + 512 * 4 + 64 * 4;
  sa2_kernel<<<16 * 128, 256, smem2, stream>>>(centers1, l1p, centers2,
                                               wt2a, bf2a, wt2b, bf2b, l2p, out_l2x);

  const size_t smem3 = (size_t)128 * 544 * 2 + (size_t)128 * 512 * 2 + 1024 * 4 + 512 * 4;
  sa3_kernel<<<16, 256, smem3, stream>>>(centers2, l2p, wt3a, bf3a, wt3b, bf3b,
                                         gh1W, gh1b, gh2W, gh2b, out_gf);

  const size_t smemp = 32 * 512 * 2 + 32 * 256 * 2;
  ph_kernel<<<64, 256, smemp, stream>>>(l2p, wtp1, bfp1, wtp2, bfp2, out_pf);
}
